// FP8SparseMoELayer_5102421148274
// MI455X (gfx1250) — matrix operand order repaired and checked by reading, compiled, not run on hardware
//
#include <hip/hip_runtime.h>
#include <hip/hip_bf16.h>

// ---------------- problem constants (match reference) ----------------
#define T_TOK 16384
#define H_DIM 1024
#define I_DIM 1024
#define E_NUM 8

// ---------------- tiling ----------------
#define BM        64                 // tokens per block tile (weight reuse)
#define XS_STRIDE (H_DIM + 8)        // LDS row stride (bf16 elems), padded
#define HS_STRIDE (I_DIM + 8)

typedef __attribute__((ext_vector_type(16))) __bf16       v16bf;
typedef __attribute__((ext_vector_type(8)))  float        v8f;
typedef __attribute__((ext_vector_type(4)))  float        v4f;
typedef __attribute__((ext_vector_type(4)))  unsigned int v4u;
typedef __attribute__((ext_vector_type(2)))  unsigned int v2u;

union ABFrag { v16bf v; v4u u[2]; };

#define V8F_ZERO {0.f, 0.f, 0.f, 0.f, 0.f, 0.f, 0.f, 0.f}

__device__ __forceinline__ void atomicAddF32(float* p, float v) {
  __hip_atomic_fetch_add(p, v, __ATOMIC_RELAXED, __HIP_MEMORY_SCOPE_AGENT);
}

// generic -> LDS(addrspace 3) byte offset, for async-to-LDS destination
typedef __attribute__((address_space(3))) unsigned char lds_byte;
__device__ __forceinline__ unsigned lds_off32(void* p) {
  return (unsigned)(unsigned long long)(lds_byte*)p;
}

// async memory -> LDS copy, 16B per lane; tracked by ASYNCcnt
__device__ __forceinline__ void async_copy_b128(unsigned lds_dst, const void* gsrc) {
  asm volatile("global_load_async_to_lds_b128 %0, %1, off"
               :: "v"(lds_dst), "v"(gsrc) : "memory");
}
__device__ __forceinline__ void wait_async0() {
  asm volatile("s_wait_asynccnt 0x0" ::: "memory");
}

// ---------------- zero output ----------------
__global__ void zero_f4_kernel(float* __restrict__ p, int n4) {
  int i = blockIdx.x * blockDim.x + threadIdx.x;
  if (i < n4) ((v4f*)p)[i] = (v4f){0.f, 0.f, 0.f, 0.f};
}

__global__ void init_cnt_kernel(int* __restrict__ cnt) {
  if (threadIdx.x < E_NUM) cnt[threadIdx.x] = 0;
}

// ---------------- fp32 -> bf16 convert (activations) ----------------
__global__ void cvt_x_kernel(const float* __restrict__ x,
                             __hip_bfloat16* __restrict__ o, int n4) {
  int i = blockIdx.x * blockDim.x + threadIdx.x;
  if (i >= n4) return;
  v4f v = ((const v4f*)x)[i];
  __hip_bfloat16 r[4];
  r[0] = __float2bfloat16(v[0]);
  r[1] = __float2bfloat16(v[1]);
  r[2] = __float2bfloat16(v[2]);
  r[3] = __float2bfloat16(v[3]);
  *(v2u*)(o + (size_t)i * 4) = *(v2u*)r;
}

// ---------------- dequant fp8-grid weights -> bf16 (lossless) --------
__global__ void dequant_kernel(const float* __restrict__ w,
                               const float* __restrict__ scale,
                               __hip_bfloat16* __restrict__ o,
                               int n4, int per_e4) {
  int i = blockIdx.x * blockDim.x + threadIdx.x;
  if (i >= n4) return;
  float s = scale[i / per_e4];
  v4f v = ((const v4f*)w)[i];
  __hip_bfloat16 r[4];
  r[0] = __float2bfloat16(v[0] * s);
  r[1] = __float2bfloat16(v[1] * s);
  r[2] = __float2bfloat16(v[2] * s);
  r[3] = __float2bfloat16(v[3] * s);
  *(v2u*)(o + (size_t)i * 4) = *(v2u*)r;
}

// ---------------- routing: softmax top-2 + expert compaction ---------
__global__ void route_kernel(const float* __restrict__ gating,
                             int* __restrict__ cnt,
                             int* __restrict__ tokL,
                             float* __restrict__ wtL) {
  int t = blockIdx.x * blockDim.x + threadIdx.x;
  if (t >= T_TOK) return;
  float g[E_NUM];
#pragma unroll
  for (int e = 0; e < E_NUM; ++e) g[e] = gating[t * E_NUM + e];
  int i1 = 0; float a1 = g[0];
#pragma unroll
  for (int e = 1; e < E_NUM; ++e) { if (g[e] > a1) { a1 = g[e]; i1 = e; } }
  int i2 = (i1 == 0) ? 1 : 0; float a2 = g[i2];
#pragma unroll
  for (int e = 0; e < E_NUM; ++e) {
    if (e != i1 && g[e] > a2) { a2 = g[e]; i2 = e; }
  }
  // renormalized top-2 softmax weights: shared denominator cancels
  float e2 = __expf(a2 - a1);
  float inv = 1.f / (1.f + e2);
  float w1v = inv;
  float w2v = e2 * inv;
  int s1 = atomicAdd(&cnt[i1], 1);
  tokL[i1 * T_TOK + s1] = t; wtL[i1 * T_TOK + s1] = w1v;
  int s2 = atomicAdd(&cnt[i2], 1);
  tokL[i2 * T_TOK + s2] = t; wtL[i2 * T_TOK + s2] = w2v;
}

// ---------------- fused expert: X@Wg^T, X@Wu^T, SwiGLU, @W2^T --------
// Block = 64 tokens of one expert. LDS holds the gathered X slab
// (64x1024 bf16, async-copied) and the SwiGLU intermediate h
// (64x1024 bf16) -> the down-proj GEMM never touches HBM for h, and
// each unique weight byte from L2 amortizes over 64 tokens.
// 16 waves; each wave owns a 32x16 output macro-tile (two 16x16 WMMA
// M sub-tiles sharing every B fragment -> 2x FLOP per loaded B byte).
// In phase 1 a wave computes BOTH gate and up accumulators for its
// macro-tile so SiLU is purely in-register.
__launch_bounds__(512)
__global__ void moe_fused_kernel(const __hip_bfloat16* __restrict__ xb,
                                 const __hip_bfloat16* __restrict__ w1b,  // [E][2I][H]
                                 const __hip_bfloat16* __restrict__ w2b,  // [E][H][I]
                                 const int*   __restrict__ cnt,
                                 const int*   __restrict__ tokL,
                                 const float* __restrict__ wtL,
                                 float* __restrict__ out) {
  extern __shared__ char smem[];
  __hip_bfloat16* Xs = (__hip_bfloat16*)smem;                                  // BM*XS_STRIDE
  __hip_bfloat16* Hs = (__hip_bfloat16*)(smem + (size_t)BM * XS_STRIDE * 2);   // BM*HS_STRIDE
  int*   sTok = (int*)(smem + (size_t)BM * XS_STRIDE * 2 + (size_t)BM * HS_STRIDE * 2);
  float* sWt  = (float*)(sTok + BM);

  const int TILES = T_TOK / BM;            // 256
  int e    = blockIdx.x / TILES;
  int tile = blockIdx.x % TILES;
  int n_e  = cnt[e];
  int row0 = tile * BM;
  if (row0 >= n_e) return;                 // uniform across block
  int rows = n_e - row0; if (rows > BM) rows = BM;

  int tid = threadIdx.x;
  if (tid < BM) {
    if (tid < rows) {
      sTok[tid] = tokL[e * T_TOK + row0 + tid];
      sWt[tid]  = wtL[e * T_TOK + row0 + tid];
    } else { sTok[tid] = 0; sWt[tid] = 0.f; }
  }
  __syncthreads();

  // gather token rows into LDS: async memory->LDS, 16B per lane-op;
  // rows past the partial tile are zero-filled with plain LDS stores.
  for (int c = tid; c < BM * (H_DIM / 8); c += 512) {
    int r = c >> 7;                 // / (H_DIM/8)
    int o = (c & 127) << 3;         // elem offset in row
    __hip_bfloat16* dst = Xs + (size_t)r * XS_STRIDE + o;
    if (r < rows) {
      async_copy_b128(lds_off32(dst), xb + (size_t)sTok[r] * H_DIM + o);
    } else {
      *(v4u*)dst = (v4u){0u, 0u, 0u, 0u};
    }
  }
  wait_async0();                    // drain this wave's ASYNCcnt
  __syncthreads();                  // publish Xs to all waves

  int wave  = tid >> 5;
  int lane  = tid & 31;
  int nt    = wave & 7;            // 8 N-tiles of 16 cols -> 128-col chunk
  int mt    = wave >> 3;           // 2 M-macros of 32 rows
  int mBase = mt * 32;
  int lhalf = lane >> 4;           // 0/1
  int lmod  = lane & 15;

  const __hip_bfloat16* w1e = w1b + (size_t)e * (2 * I_DIM) * H_DIM;
  const __hip_bfloat16* w2e = w2b + (size_t)e * H_DIM * I_DIM;

  const int ak = lhalf * 8;        // A frag: 16B chunks at k+ak and k+ak+16
  const int bk = lhalf * 8;        // B frag: 16B chunks at k+bk and k+bk+16
  const __hip_bfloat16* aRow0 = Xs + (size_t)(mBase + lmod) * XS_STRIDE;
  const __hip_bfloat16* aRow1 = Xs + (size_t)(mBase + 16 + lmod) * XS_STRIDE;

  // ---------------- phase 1: gate/up GEMMs + SwiGLU -> Hs ----------------
  for (int nc = 0; nc < I_DIM / 128; ++nc) {
    int n0 = nc * 128 + nt * 16;   // intermediate channel base of this tile
    v8f accg0 = V8F_ZERO, accg1 = V8F_ZERO;
    v8f accu0 = V8F_ZERO, accu1 = V8F_ZERO;
    const __hip_bfloat16* bgRow = w1e + (size_t)(n0 + lmod) * H_DIM;           // gate row n
    const __hip_bfloat16* buRow = w1e + (size_t)(I_DIM + n0 + lmod) * H_DIM;   // up row n
#pragma unroll 2
    for (int kk = 0; kk < H_DIM; kk += 32) {
      ABFrag a0, a1, bg, bu;
      a0.u[0] = *(const v4u*)(aRow0 + kk + ak);
      a0.u[1] = *(const v4u*)(aRow0 + kk + ak + 16);
      a1.u[0] = *(const v4u*)(aRow1 + kk + ak);
      a1.u[1] = *(const v4u*)(aRow1 + kk + ak + 16);
      bg.u[0] = *(const v4u*)(bgRow + kk + bk);
      bg.u[1] = *(const v4u*)(bgRow + kk + bk + 16);
      bu.u[0] = *(const v4u*)(buRow + kk + bk);
      bu.u[1] = *(const v4u*)(buRow + kk + bk + 16);
      accg0 = __builtin_amdgcn_wmma_f32_16x16x32_bf16(false, a0.v, false, bg.v,
                                                      (short)0, accg0, false, false);
      accu0 = __builtin_amdgcn_wmma_f32_16x16x32_bf16(false, a0.v, false, bu.v,
                                                      (short)0, accu0, false, false);
      accg1 = __builtin_amdgcn_wmma_f32_16x16x32_bf16(false, a1.v, false, bg.v,
                                                      (short)0, accg1, false, false);
      accu1 = __builtin_amdgcn_wmma_f32_16x16x32_bf16(false, a1.v, false, bu.v,
                                                      (short)0, accu1, false, false);
    }
    // SwiGLU in-register, write bf16 h tiles to LDS
#pragma unroll
    for (int j = 0; j < 8; ++j) {
      int mr = j + lhalf * 8;                   // C layout: VGPR j, lane half
      float g0 = accg0[j], u0 = accu0[j];
      float h0 = (g0 / (1.f + __expf(-g0))) * u0;
      Hs[(size_t)(mBase + mr) * HS_STRIDE + n0 + lmod] = __float2bfloat16(h0);
      float g1 = accg1[j], u1 = accu1[j];
      float h1 = (g1 / (1.f + __expf(-g1))) * u1;
      Hs[(size_t)(mBase + 16 + mr) * HS_STRIDE + n0 + lmod] = __float2bfloat16(h1);
    }
  }
  __syncthreads();

  // ---------------- phase 2: h @ W2^T, scatter-add into out ----------------
  const __hip_bfloat16* hRow0 = Hs + (size_t)(mBase + lmod) * HS_STRIDE;
  const __hip_bfloat16* hRow1 = Hs + (size_t)(mBase + 16 + lmod) * HS_STRIDE;
  for (int oc = 0; oc < H_DIM / 128; ++oc) {
    int n0 = oc * 128 + nt * 16;   // output channel base of this tile
    v8f acc0 = V8F_ZERO, acc1 = V8F_ZERO;
    const __hip_bfloat16* bRow = w2e + (size_t)(n0 + lmod) * I_DIM;  // W2 row n
#pragma unroll 4
    for (int kk = 0; kk < I_DIM; kk += 32) {
      ABFrag a0, a1, b;
      a0.u[0] = *(const v4u*)(hRow0 + kk + ak);
      a0.u[1] = *(const v4u*)(hRow0 + kk + ak + 16);
      a1.u[0] = *(const v4u*)(hRow1 + kk + ak);
      a1.u[1] = *(const v4u*)(hRow1 + kk + ak + 16);
      b.u[0]  = *(const v4u*)(bRow + kk + bk);
      b.u[1]  = *(const v4u*)(bRow + kk + bk + 16);
      acc0 = __builtin_amdgcn_wmma_f32_16x16x32_bf16(false, a0.v, false, b.v,
                                                     (short)0, acc0, false, false);
      acc1 = __builtin_amdgcn_wmma_f32_16x16x32_bf16(false, a1.v, false, b.v,
                                                     (short)0, acc1, false, false);
    }
#pragma unroll
    for (int j = 0; j < 8; ++j) {
      int mr = j + lhalf * 8;
      int mrow0 = mBase + mr;
      if (mrow0 < rows) {
        atomicAddF32(out + (size_t)sTok[mrow0] * H_DIM + n0 + lmod,
                     sWt[mrow0] * acc0[j]);
      }
      int mrow1 = mBase + 16 + mr;
      if (mrow1 < rows) {
        atomicAddF32(out + (size_t)sTok[mrow1] * H_DIM + n0 + lmod,
                     sWt[mrow1] * acc1[j]);
      }
    }
  }
}

// ---------------- host launcher ----------------
extern "C" void kernel_launch(void* const* d_in, const int* in_sizes, int n_in,
                              void* d_out, int out_size, void* d_ws, size_t ws_size,
                              hipStream_t stream) {
  const float* x      = (const float*)d_in[0];
  const float* gating = (const float*)d_in[1];
  const float* w1     = (const float*)d_in[2];
  const float* w2     = (const float*)d_in[3];
  const float* w1s    = (const float*)d_in[4];
  const float* w2s    = (const float*)d_in[5];
  float* out = (float*)d_out;

  char* ws = (char*)d_ws;
  size_t off = 0;
  __hip_bfloat16* xb  = (__hip_bfloat16*)(ws + off); off += (size_t)T_TOK * H_DIM * 2;
  __hip_bfloat16* w1b = (__hip_bfloat16*)(ws + off); off += (size_t)E_NUM * 2 * I_DIM * H_DIM * 2;
  __hip_bfloat16* w2b = (__hip_bfloat16*)(ws + off); off += (size_t)E_NUM * H_DIM * I_DIM * 2;
  int*   cnt  = (int*)(ws + off);   off += 64;
  int*   tokL = (int*)(ws + off);   off += (size_t)E_NUM * T_TOK * 4;
  float* wtL  = (float*)(ws + off); off += (size_t)E_NUM * T_TOK * 4;

  // 1) zero output (atomic scatter-add target) and expert counters
  {
    int n4 = T_TOK * H_DIM / 4;
    zero_f4_kernel<<<(n4 + 255) / 256, 256, 0, stream>>>(out, n4);
    init_cnt_kernel<<<1, 32, 0, stream>>>(cnt);
  }
  // 2) activations fp32 -> bf16
  {
    int n4 = T_TOK * H_DIM / 4;
    cvt_x_kernel<<<(n4 + 255) / 256, 256, 0, stream>>>(x, xb, n4);
  }
  // 3) dequant weights (fp8-grid * scale, exact in bf16)
  {
    int n4 = E_NUM * 2 * I_DIM * H_DIM / 4;
    dequant_kernel<<<(n4 + 255) / 256, 256, 0, stream>>>(w1, w1s, w1b, n4,
                                                         2 * I_DIM * H_DIM / 4);
  }
  {
    int n4 = E_NUM * H_DIM * I_DIM / 4;
    dequant_kernel<<<(n4 + 255) / 256, 256, 0, stream>>>(w2, w2s, w2b, n4,
                                                         H_DIM * I_DIM / 4);
  }
  // 4) routing + compaction
  route_kernel<<<(T_TOK + 255) / 256, 256, 0, stream>>>(gating, cnt, tokL, wtL);

  // 5) fused expert GEMMs (WMMA bf16, f32 accum), scatter-add to out
  {
    size_t smem = (size_t)BM * XS_STRIDE * 2   // Xs
                + (size_t)BM * HS_STRIDE * 2   // Hs
                + (size_t)BM * 4               // sTok
                + (size_t)BM * 4;              // sWt
    dim3 grid(E_NUM * (T_TOK / BM));
    moe_fused_kernel<<<grid, 512, smem, stream>>>(xb, w1b, w2b, cnt, tokL, wtL, out);
  }
}
